// MAFLayer_16810501997248
// MI455X (gfx1250) — matrix operand order repaired and checked by reading, not compiled, not run on hardware
//
#include <hip/hip_runtime.h>
#include <hip/hip_bf16.h>

#define DIM   64
#define H     1024
#define BATCH 32768
#define MT    64          // batch rows per workgroup
#define SST   1032        // LDS row stride for h1/h2 (bf16 elems, padded)
#define XS    72          // LDS row stride for x_bf16

typedef __attribute__((ext_vector_type(16))) __bf16 v16bf;
typedef __attribute__((ext_vector_type(8)))  __bf16 v8bf;
typedef __attribute__((ext_vector_type(8)))  float  v8f;

union AFrag { unsigned int u[8]; v16bf v; };
union BFrag { v8bf h[2];         v16bf v; };

// ---------------- prep: mask + bf16 + B-fragment swizzle ----------------
// B fragment layout (K=32 x N=16 tile, 512 bf16):
//   offset = load*256 + lane*8 + 2*v2 + e ; v = load*4+v2
//   n = nt*16 + (lane&15) ; k = kt*32 + 2*v + 16*(lane>>4) + e
__global__ __launch_bounds__(256) void prep_w1(const float* __restrict__ W1,
                                               __bf16* __restrict__ w1f) {
  int idx  = blockIdx.x * 256 + threadIdx.x;          // 2*64*512 = 65536
  int tile = idx >> 9, r = idx & 511;
  int kt = tile & 1, nt = tile >> 1;
  int load = r >> 8, lane = (r >> 3) & 31, j = r & 7;
  int v = load * 4 + (j >> 1), e = j & 1;
  int n = nt * 16 + (lane & 15);
  int k = kt * 32 + 2 * v + 16 * (lane >> 4) + e;
  float m = ((n % 63) >= k) ? 1.f : 0.f;              // M1: deg_h1[n] >= deg_in[k]
  w1f[idx] = (__bf16)(W1[k * H + n] * m);
}

__global__ __launch_bounds__(256) void prep_w2(const float* __restrict__ W2,
                                               __bf16* __restrict__ w2f) {
  int idx  = blockIdx.x * 256 + threadIdx.x;          // 32*64*512 = 1048576
  int tile = idx >> 9, r = idx & 511;
  int kt = tile & 31, nt = tile >> 5;
  int load = r >> 8, lane = (r >> 3) & 31, j = r & 7;
  int v = load * 4 + (j >> 1), e = j & 1;
  int n = nt * 16 + (lane & 15);
  int k = kt * 32 + 2 * v + 16 * (lane >> 4) + e;
  float m = ((n % 63) >= (k % 63)) ? 1.f : 0.f;       // M2
  w2f[idx] = (__bf16)(W2[k * H + n] * m);
}

// W3 [1024, 128] -> bf16 B fragments, tiles (nt 0..7, kt 0..31), M3 pre-applied
__global__ __launch_bounds__(256) void prep_w3(const float* __restrict__ W3,
                                               __bf16* __restrict__ w3f) {
  int idx  = blockIdx.x * 256 + threadIdx.x;          // 8*32*512 = 131072
  int tile = idx >> 9, r = idx & 511;
  int kt = tile & 31, nt = tile >> 5;
  int load = r >> 8, lane = (r >> 3) & 31, j = r & 7;
  int v = load * 4 + (j >> 1), e = j & 1;
  int n = nt * 16 + (lane & 15);
  int k = kt * 32 + 2 * v + 16 * (lane >> 4) + e;
  float m = ((n & 63) > (k % 63)) ? 1.f : 0.f;        // M3: deg_out[n] > deg_h2[k]
  w3f[idx] = (__bf16)(W3[k * (2 * DIM) + n] * m);
}

// ---------------- main: persistent autoregressive tile kernel ----------------
__global__ __launch_bounds__(256) void maf_kernel(
    const float* __restrict__ u,  const float* __restrict__ b1,
    const float* __restrict__ b2, const float* __restrict__ b3,
    const __bf16* __restrict__ w1f, const __bf16* __restrict__ w2f,
    const __bf16* __restrict__ w3f,
    float* __restrict__ out_x, float* __restrict__ out_logd) {
  __shared__ __bf16 sh1[MT * SST];       // 129 KB
  __shared__ __bf16 sh2[MT * SST];       // 129 KB
  __shared__ float  sx [MT * DIM];       // 16 KB
  __shared__ float  su [MT * DIM];       // 16 KB
  __shared__ __bf16 sxb[MT * XS];        // 9 KB
  __shared__ float  smu[MT];
  __shared__ float  sls[MT];

  const int tid  = threadIdx.x;
  const int lane = tid & 31, w = tid >> 5;
  const int half = lane >> 4, mrow = lane & 15;
  const int wgbase = blockIdx.x * MT;

  for (int i = tid; i < MT * DIM; i += 256) {
    sx[i] = 0.f;
    su[i] = u[(size_t)wgbase * DIM + i];             // coalesced one-time stage
  }
  for (int i = tid; i < MT * XS; i += 256) sxb[i] = (__bf16)0.f;
  float logd = 0.f;
  __syncthreads();

  for (int t = 0; t < DIM; ++t) {
    // ---- GEMM1: h1 = relu(x @ W1m + b1) -------------------------------
    for (int mt = 0; mt < 4; ++mt) {
      for (int ntl = 0; ntl < 8; ++ntl) {
        int nt = w * 8 + ntl;
        v8f acc = {};
#pragma unroll
        for (int kt = 0; kt < 2; ++kt) {
          AFrag a;
#pragma unroll
          for (int v = 0; v < 8; ++v) {
            int kp = kt * 32 + 2 * v + 16 * half;
            a.u[v] = *(const unsigned int*)&sxb[(mt * 16 + mrow) * XS + kp];
          }
          BFrag b;
          const v8bf* bp = (const v8bf*)w1f + (unsigned)(nt * 2 + kt) * 64;
          b.h[0] = bp[lane];
          b.h[1] = bp[32 + lane];
          acc = __builtin_amdgcn_wmma_f32_16x16x32_bf16(false, a.v, false, b.v,
                                                        (short)0, acc, false, false);
        }
        int ncol = nt * 16 + mrow;
        float bias = b1[ncol];
#pragma unroll
        for (int r = 0; r < 8; ++r) {
          float f = acc[r] + bias;
          sh1[(mt * 16 + r + 8 * half) * SST + ncol] = (__bf16)(f > 0.f ? f : 0.f);
        }
      }
    }
    __syncthreads();

    // ---- GEMM2: h2 = relu(h1 @ W2m + b2), NT-unroll 4 -----------------
    for (int mt = 0; mt < 4; ++mt) {
      for (int g = 0; g < 2; ++g) {
        int nt0 = w * 8 + g * 4;
        v8f acc0 = {}, acc1 = {}, acc2 = {}, acc3 = {};
        for (int kt = 0; kt < 32; ++kt) {
          AFrag a;
#pragma unroll
          for (int v = 0; v < 8; ++v) {
            int kp = kt * 32 + 2 * v + 16 * half;
            a.u[v] = *(const unsigned int*)&sh1[(mt * 16 + mrow) * SST + kp];
          }
#pragma unroll
          for (int q = 0; q < 4; ++q) {
            BFrag b;
            const v8bf* bp = (const v8bf*)w2f + (unsigned)((nt0 + q) * 32 + kt) * 64;
            b.h[0] = bp[lane];
            b.h[1] = bp[32 + lane];
            v8f& A = (q == 0 ? acc0 : q == 1 ? acc1 : q == 2 ? acc2 : acc3);
            A = __builtin_amdgcn_wmma_f32_16x16x32_bf16(false, a.v, false, b.v,
                                                        (short)0, A, false, false);
          }
        }
#pragma unroll
        for (int q = 0; q < 4; ++q) {
          int ncol = (nt0 + q) * 16 + mrow;
          float bias = b2[ncol];
          v8f A = (q == 0 ? acc0 : q == 1 ? acc1 : q == 2 ? acc2 : acc3);
#pragma unroll
          for (int r = 0; r < 8; ++r) {
            float f = A[r] + bias;
            sh2[(mt * 16 + r + 8 * half) * SST + ncol] = (__bf16)(f > 0.f ? f : 0.f);
          }
        }
      }
    }
    __syncthreads();

    // ---- GEMM3 (WMMA): only the 16-col tiles holding cols t and 64+t --
    {
      int wsel = w >> 2;                 // 0 -> mu tile, 1 -> log_sigma tile
      int mt   = w & 3;
      int nt   = wsel * 4 + (t >> 4);
      v8f acc = {};
      for (int kt = 0; kt < 32; ++kt) {
        AFrag a;
#pragma unroll
        for (int v = 0; v < 8; ++v) {
          int kp = kt * 32 + 2 * v + 16 * half;
          a.u[v] = *(const unsigned int*)&sh2[(mt * 16 + mrow) * SST + kp];
        }
        BFrag b;
        const v8bf* bp = (const v8bf*)w3f + (unsigned)(nt * 32 + kt) * 64;
        b.h[0] = bp[lane];
        b.h[1] = bp[32 + lane];
        acc = __builtin_amdgcn_wmma_f32_16x16x32_bf16(false, a.v, false, b.v,
                                                      (short)0, acc, false, false);
      }
      // column t (resp. 64+t) lives in lanes with mrow == t%16
      if (mrow == (t & 15)) {
        float* dst = wsel ? sls : smu;
#pragma unroll
        for (int r = 0; r < 8; ++r) dst[mt * 16 + r + 8 * half] = acc[r];
      }
    }
    __syncthreads();

    // ---- coupling update: x[:,t] = u[:,t]*exp(ls) + mu ----------------
    if (tid < MT) {
      int r = tid;
      float mu = smu[r] + b3[t];
      float ls = sls[r] + b3[DIM + t];
      float xn = fmaf(su[r * DIM + t], __expf(ls), mu);
      sx [r * DIM + t] = xn;
      sxb[r * XS  + t] = (__bf16)xn;
      logd += ls;
    }
    __syncthreads();
  }

  for (int i = tid; i < MT * DIM; i += 256) {
    int r = i / DIM, c = i % DIM;
    out_x[(size_t)(wgbase + r) * DIM + c] = sx[i];
  }
  if (tid < MT) out_logd[wgbase + tid] = logd;
}

// ---------------- host entry ----------------
extern "C" void kernel_launch(void* const* d_in, const int* in_sizes, int n_in,
                              void* d_out, int out_size, void* d_ws, size_t ws_size,
                              hipStream_t stream) {
  const float* u  = (const float*)d_in[0];
  const float* W1 = (const float*)d_in[1];
  const float* b1 = (const float*)d_in[2];
  const float* W2 = (const float*)d_in[3];
  const float* b2 = (const float*)d_in[4];
  const float* W3 = (const float*)d_in[5];
  const float* b3 = (const float*)d_in[6];

  char* ws = (char*)d_ws;
  __bf16* w1f = (__bf16*)ws;                              // 65536   bf16 = 128 KB
  __bf16* w2f = (__bf16*)(ws + 131072);                   // 1048576 bf16 = 2 MB
  __bf16* w3f = (__bf16*)(ws + 131072 + 2097152);         // 131072  bf16 = 256 KB

  prep_w1<<<65536 / 256, 256, 0, stream>>>(W1, w1f);
  prep_w2<<<1048576 / 256, 256, 0, stream>>>(W2, w2f);
  prep_w3<<<131072 / 256, 256, 0, stream>>>(W3, w3f);

  float* out_x    = (float*)d_out;
  float* out_logd = out_x + (size_t)BATCH * DIM;
  maf_kernel<<<BATCH / MT, 256, 0, stream>>>(u, b1, b2, b3, w1f, w2f, w3f,
                                             out_x, out_logd);
}
